// IPEXEmptyINT4Linear_16801912062544
// MI455X (gfx1250) — compile-verified, matrix operand order checked by reading, not run on hardware
//
#include <hip/hip_runtime.h>

typedef __attribute__((ext_vector_type(16))) _Float16 v16h;
typedef __attribute__((ext_vector_type(8)))  _Float16 v8h;
typedef __attribute__((ext_vector_type(4)))  _Float16 v4h;
typedef __attribute__((ext_vector_type(8)))  float    v8f;

#define BM 128
#define BN 128
#define BK 32
#define LDA 40          // half stride (80B = 5*16B: aligned b128 slots, bank-skewed)
#define LDB 40
#define ABUF (BM * LDA)
#define BBUF (BN * LDB)
#define BUFSZ (ABUF + BBUF)

__global__ __launch_bounds__(256)
void int4_gemm_wmma_f16(const float* __restrict__ A,
                        const int*   __restrict__ QW,
                        const float* __restrict__ SC,
                        const int*   __restrict__ QZ,
                        const float* __restrict__ BIAS,
                        float* __restrict__ C,
                        int M, int N, int K, int lgG)
{
    __shared__ _Float16 lds[2 * BUFSZ];     // ping-pong [A|B] tiles, 40KB

    const int tid  = threadIdx.x;
    const int lane = tid & 31;
    const int wave = tid >> 5;
    const int wm   = (wave >> 2) * 64;
    const int wn   = (wave & 3) * 32;
    const int bm   = blockIdx.y * BM;
    const int bn   = blockIdx.x * BN;
    const int N8   = N >> 3;
    const int nT   = K / BK;

    // ---- streaming per-thread pointers (point at tile 0) ----
    const int r0 = tid >> 3;                 // A row 0..31 (+32*i)
    const int c4 = tid & 7;                  // float4 column within 32-K slab
    const float* aCur = A + (size_t)(bm + r0) * K + c4 * 4;
    const size_t aRow = (size_t)32 * K;      // invariant row-block stride (imm24-friendly)

    const int nl  = tid & 127;               // B column 0..127 (same for both qw words)
    const int kr0 = tid >> 7;                // qweight row 0..1 (+2 for second word)
    const int nq  = bn + nl;
    const int* qCur = QW + (size_t)kr0 * N + nq;
    const size_t qOff2  = (size_t)2 * N;     // second qweight word (imm offset)
    const size_t qStep  = (size_t)(BK >> 3) * N;   // slab advance: 4N ints

    v8f acc[4][2];
#pragma unroll
    for (int i = 0; i < 4; ++i)
#pragma unroll
        for (int j = 0; j < 2; ++j)
            acc[i][j] = (v8f)0.0f;

    // in-flight staging registers
    float4 aReg[4];
    int    qwReg[2];
    float  scReg, zsReg;

    auto GLOAD = [&](int kNext) {
#pragma unroll
        for (int i = 0; i < 4; ++i)
            aReg[i] = *(const float4*)(aCur + (size_t)i * aRow);   // base + imm24
        qwReg[0] = qCur[0];
        qwReg[1] = qCur[qOff2];
        const int   g  = kNext >> lgG;                 // shift, not divide
        const float sc = SC[(size_t)g * N + nq];
        const int   zw = QZ[(size_t)g * N8 + (nq >> 3)];
        const int   zp = (zw >> ((nq & 7) * 4)) & 0xF;
        scReg = sc;
        zsReg = (float)zp * sc;
    };

    auto LSTORE = [&](_Float16* bufA, _Float16* bufB) {
#pragma unroll
        for (int i = 0; i < 4; ++i) {
            v4h hv = { (_Float16)aReg[i].x, (_Float16)aReg[i].y,
                       (_Float16)aReg[i].z, (_Float16)aReg[i].w };
            *(v4h*)(&bufA[(r0 + 32 * i) * LDA + c4 * 4]) = hv;     // ds_store_b64
        }
#pragma unroll
        for (int i = 0; i < 2; ++i) {
            v8h hv;
#pragma unroll
            for (int j = 0; j < 8; ++j) {
                const int q = (qwReg[i] >> (4 * j)) & 0xF;
                hv[j] = (_Float16)(fmaf((float)q, scReg, -zsReg));
            }
            *(v8h*)(&bufB[nl * LDB + (kr0 + 2 * i) * 8]) = hv;     // ds_store_b128
        }
    };

    auto COMPUTE = [&](const _Float16* bufA, const _Float16* bufB) {
        const int koff = (lane < 16) ? 0 : 8;   // ISA 16-bit fragment K-split
        const int rsel = lane & 15;

        v16h bfrag[2], afrag[4];
#pragma unroll
        for (int j = 0; j < 2; ++j) {
            const _Float16* p = &bufB[(wn + j * 16 + rsel) * LDB];
            v8h lo = *(const v8h*)(p + koff);
            v8h hi = *(const v8h*)(p + koff + 16);
            bfrag[j] = __builtin_shufflevector(lo, hi, 0,1,2,3,4,5,6,7,8,9,10,11,12,13,14,15);
        }
#pragma unroll
        for (int i = 0; i < 4; ++i) {
            const _Float16* p = &bufA[(wm + i * 16 + rsel) * LDA];
            v8h lo = *(const v8h*)(p + koff);
            v8h hi = *(const v8h*)(p + koff + 16);
            afrag[i] = __builtin_shufflevector(lo, hi, 0,1,2,3,4,5,6,7,8,9,10,11,12,13,14,15);
        }
#pragma unroll
        for (int i = 0; i < 4; ++i)
#pragma unroll
            for (int j = 0; j < 2; ++j)
                acc[i][j] = __builtin_amdgcn_wmma_f32_16x16x32_f16(
                    false, afrag[i], false, bfrag[j],
                    (short)0, acc[i][j], false, false);
    };

    // ---- prologue: stage tile 0, advance to tile 1 ----
    int kNext = 0;
    GLOAD(0);
    {
        const int more = (nT > 1);
        aCur  += more ? BK : 0;
        qCur  += more ? qStep : 0;
        kNext  = more ? BK : 0;
    }
    LSTORE(lds, lds + ABUF);
    __syncthreads();

    // ---- steady state: loads -> fence -> WMMA -> fence -> cvt+store, 1 barrier ----
    for (int t = 0; t < nT; ++t) {
        _Float16* curT = lds + (t & 1) * BUFSZ;
        _Float16* nxtT = lds + ((t & 1) ^ 1) * BUFSZ;

        // L2 warm-up one slab beyond the in-flight one (speculative)
        __builtin_prefetch(aCur + BK, 0, 2);
        __builtin_prefetch(qCur + qStep, 0, 2);

        GLOAD(kNext);                          // all next-tile loads issued, no waits
        __builtin_amdgcn_sched_barrier(0);     // keep them above the WMMA block
        COMPUTE(curT, curT + ABUF);
        __builtin_amdgcn_sched_barrier(0);     // keep converts/stores below WMMAs
        LSTORE(nxtT, nxtT + ABUF);

        const int more = (t + 2 < nT);         // branchless tail (redundant last reload)
        aCur  += more ? BK : 0;
        qCur  += more ? qStep : 0;
        kNext += more ? BK : 0;
        __syncthreads();
    }

    // ---- epilogue: D layout (VGPR r: lanes0-15 -> M=r, lanes16-31 -> M=r+8) ----
    const int rowoff = (lane < 16) ? 0 : 8;
    const int csel   = lane & 15;
#pragma unroll
    for (int i = 0; i < 4; ++i) {
#pragma unroll
        for (int j = 0; j < 2; ++j) {
            const int col = bn + wn + j * 16 + csel;
            const float b = BIAS[col];
#pragma unroll
            for (int r = 0; r < 8; ++r) {
                const int row = bm + wm + i * 16 + rowoff + r;
                C[(size_t)row * N + col] = acc[i][j][r] + b;
            }
        }
    }
}

extern "C" void kernel_launch(void* const* d_in, const int* in_sizes, int n_in,
                              void* d_out, int out_size, void* d_ws, size_t ws_size,
                              hipStream_t stream) {
    (void)n_in; (void)d_ws; (void)ws_size; (void)out_size;
    const float* A    = (const float*)d_in[0];
    const int*   QW   = (const int*)  d_in[1];
    const float* SC   = (const float*)d_in[2];
    const int*   QZ   = (const int*)  d_in[3];
    const float* BIAS = (const float*)d_in[4];
    float* C = (float*)d_out;

    const int N   = in_sizes[4];
    const int K   = (int)((long long)in_sizes[1] * 8 / N);
    const int M   = (int)((long long)in_sizes[0] / K);
    const int NG  = (int)((long long)in_sizes[2] / N);
    const int G   = K / NG;
    const int lgG = 31 - __builtin_clz((unsigned)G);   // G is a power of two (128)

    dim3 grid((N + BN - 1) / BN, (M + BM - 1) / BM);
    dim3 block(256);
    int4_gemm_wmma_f16<<<grid, block, 0, stream>>>(A, QW, SC, QZ, BIAS, C, M, N, K, lgG);
}
